// AttentionBlock_12584254177821
// MI455X (gfx1250) — matrix operand order repaired and checked by reading, not compiled, not run on hardware
//
#include <hip/hip_runtime.h>
#include <hip/hip_bf16.h>

// ---------------------------------------------------------------------------
// Types for CDNA5 WMMA (wave32): bf16 A/B fragments (16 bf16/lane = 8 VGPRs),
// fp32 C/D fragment (8 f32/lane = 8 VGPRs).
// ---------------------------------------------------------------------------
typedef __attribute__((ext_vector_type(16))) __bf16 bf16x16;
typedef __attribute__((ext_vector_type(8)))  __bf16 bf16x8;
typedef __attribute__((ext_vector_type(8)))  float  f32x8;

#define BB 2
#define TT 2048
#define DD 768
#define HH 12
#define HSZ 64
#define FFD 3072
#define HALFD 1536
#define BT (BB*TT)

__device__ __forceinline__ __bf16 to_bf16(float f) {
  unsigned int u = __builtin_bit_cast(unsigned int, f);
  u += 0x7FFFu + ((u >> 16) & 1u);          // round-to-nearest-even
  unsigned short h = (unsigned short)(u >> 16);
  return __builtin_bit_cast(__bf16, h);
}

// A-fragment (16x32, MxK) from row-major bf16 [rows, ld]:
// lane m<16 holds row m, K = [k0..k0+16) with k0 = (lane>>4)*16
__device__ __forceinline__ bf16x16 load_frag_a(const __bf16* src, int ld, int lane) {
  int row = lane & 15;
  int k0  = (lane >> 4) * 16;
  const __bf16* p = src + (size_t)row * ld + k0;
  bf16x8 lo = *(const bf16x8*)(p);
  bf16x8 hi = *(const bf16x8*)(p + 8);
  bf16x16 r;
#pragma unroll
  for (int i = 0; i < 8; ++i) { r[i] = lo[i]; r[i + 8] = hi[i]; }
  return r;
}

// B-fragment (32x16, KxN) from B^T storage (row n holds K values contiguously):
// lane n<16 holds N=n, K=[0..16); lane n+16 holds N=n, K=[16..32)
__device__ __forceinline__ bf16x16 load_frag_b(const __bf16* bt, int ld, int lane) {
  int n  = lane & 15;
  int k0 = (lane >> 4) * 16;
  return *(const bf16x16*)(bt + (size_t)n * ld + k0);
}

__device__ __forceinline__ f32x8 wmma_bf16(bf16x16 a, bf16x16 b, f32x8 c) {
  // (neg_a, A, neg_b, B, c_mod, C, reuse_a, reuse_b)
  return __builtin_amdgcn_wmma_f32_16x16x32_bf16(false, a, false, b, (short)0, c,
                                                 false, false);
}

// ---------------------------------------------------------------------------
// Weight fp32 -> bf16 transposed casts
// ---------------------------------------------------------------------------
// Wq/Wk/Wv: [H][D][HS] fp32 -> [H*HS][D] bf16 (B^T layout: out[n][d] = in[h][d][j])
__global__ void cast_T_qkv_kernel(const float* __restrict__ in, __bf16* __restrict__ out) {
  int idx = blockIdx.x * blockDim.x + threadIdx.x;
  if (idx >= DD * DD) return;
  int n = idx / DD, d = idx % DD;
  int h = n / HSZ, j = n % HSZ;
  out[idx] = to_bf16(in[((size_t)h * DD + d) * HSZ + j]);
}

// generic [K][N] fp32 -> [N][K] bf16
__global__ void cast_T_2d_kernel(const float* __restrict__ in, __bf16* __restrict__ out,
                                 int K, int N) {
  long long idx = (long long)blockIdx.x * blockDim.x + threadIdx.x;
  if (idx >= (long long)K * N) return;
  int n = (int)(idx / K), k = (int)(idx % K);
  out[idx] = to_bf16(in[(size_t)k * N + n]);
}

// ---------------------------------------------------------------------------
// RMSNorm (p = -1 style): scale * x / (sqrt(mean(x^2)) + eps)  -> bf16
// ---------------------------------------------------------------------------
__global__ void rmsnorm_cast_kernel(const float* __restrict__ x,
                                    const float* __restrict__ g,
                                    __bf16* __restrict__ out) {
  int row = blockIdx.x;
  const float* xr = x + (size_t)row * DD;
  __shared__ float red[256];
  float ss = 0.f;
  for (int i = threadIdx.x; i < DD; i += 256) { float v = xr[i]; ss += v * v; }
  red[threadIdx.x] = ss;
  __syncthreads();
  for (int off = 128; off > 0; off >>= 1) {
    if ((int)threadIdx.x < off) red[threadIdx.x] += red[threadIdx.x + off];
    __syncthreads();
  }
  float rms = sqrtf(red[0] * (1.0f / DD));
  float inv = 1.0f / (rms + 1e-8f);
  for (int i = threadIdx.x; i < DD; i += 256)
    out[(size_t)row * DD + i] = to_bf16(xr[i] * inv * g[i]);
}

// ---------------------------------------------------------------------------
// QKV projections: one wave = 16(M) x 64(N) tile; grid.z selects Q/K/V.
// Q,K written as [b][h][T][64]; V written transposed [b][h][64][T].
// ---------------------------------------------------------------------------
__global__ void qkv_kernel(const __bf16* __restrict__ Hbf,
                           const __bf16* __restrict__ WqT,
                           const __bf16* __restrict__ WkT,
                           const __bf16* __restrict__ WvT,
                           __bf16* __restrict__ Q, __bf16* __restrict__ Ko,
                           __bf16* __restrict__ V) {
  int lane = threadIdx.x;
  int which = blockIdx.z;
  const __bf16* Wt = (which == 0) ? WqT : (which == 1) ? WkT : WvT;
  int n0 = blockIdx.x * 64, m0 = blockIdx.y * 16;
  f32x8 acc[4] = {};
  for (int k0 = 0; k0 < DD; k0 += 32) {
    bf16x16 a = load_frag_a(Hbf + (size_t)m0 * DD + k0, DD, lane);
#pragma unroll
    for (int j = 0; j < 4; ++j) {
      bf16x16 b = load_frag_b(Wt + (size_t)(n0 + 16 * j) * DD + k0, DD, lane);
      acc[j] = wmma_bf16(a, b, acc[j]);
    }
  }
  int g = lane >> 4, c = lane & 15;
#pragma unroll
  for (int j = 0; j < 4; ++j) {
#pragma unroll
    for (int r = 0; r < 8; ++r) {
      int row = m0 + r + 8 * g;               // global token index
      int b = row >> 11, t = row & (TT - 1);
      int col = n0 + 16 * j + c;
      int h = col >> 6, hs = col & 63;
      __bf16 val = to_bf16(acc[j][r]);
      size_t bh = (size_t)b * HH + h;
      if (which == 0)      Q[(bh * TT + t) * HSZ + hs] = val;
      else if (which == 1) Ko[(bh * TT + t) * HSZ + hs] = val;
      else                 V[(bh * HSZ + hs) * TT + t] = val;
    }
  }
}

// ---------------------------------------------------------------------------
// Flash-style causal attention. One wave per (b,h, 16-row tile).
// scores = (q @ k^T) * D^-0.5, online softmax in fp32, P@V via WMMA,
// P re-fragmented through LDS. Output: attn bf16 [B*T][D].
// ---------------------------------------------------------------------------
__global__ void attention_kernel(const __bf16* __restrict__ Q,
                                 const __bf16* __restrict__ Kt,
                                 const __bf16* __restrict__ Vt,
                                 __bf16* __restrict__ attn_out) {
  const float SCALE = 0.03608439182435161f;   // 768^-0.5
  const float NEG   = -3.0e38f;
  int lane = threadIdx.x;
  int t0 = blockIdx.x * 16;
  int bh = blockIdx.y;
  int b = bh / HH, h = bh % HH;
  const __bf16* qh = Q  + (size_t)bh * TT * HSZ;   // [T][64]
  const __bf16* kh = Kt + (size_t)bh * TT * HSZ;   // [T][64]
  const __bf16* vh = Vt + (size_t)bh * HSZ * TT;   // [64][T]
  __shared__ __align__(32) __bf16 Plds[16 * 32];

  bf16x16 qa0 = load_frag_a(qh + (size_t)t0 * HSZ + 0,  HSZ, lane);
  bf16x16 qa1 = load_frag_a(qh + (size_t)t0 * HSZ + 32, HSZ, lane);

  f32x8 acc[4] = {};
  float mrun[8], lrun[8];
#pragma unroll
  for (int r = 0; r < 8; ++r) { mrun[r] = NEG; lrun[r] = 0.f; }
  int g = lane >> 4, c = lane & 15;

  for (int s0 = 0; s0 <= t0 + 15; s0 += 32) {
    f32x8 sc[2];
#pragma unroll
    for (int half = 0; half < 2; ++half) {
      int n0 = s0 + 16 * half;
      bf16x16 kb0 = load_frag_b(kh + (size_t)n0 * HSZ + 0,  HSZ, lane);
      bf16x16 kb1 = load_frag_b(kh + (size_t)n0 * HSZ + 32, HSZ, lane);
      f32x8 s = {};
      s = wmma_bf16(qa0, kb0, s);
      s = wmma_bf16(qa1, kb1, s);
#pragma unroll
      for (int r = 0; r < 8; ++r) {
        int trow = t0 + r + 8 * g;
        int scol = n0 + c;
        float v = s[r] * SCALE;
        if (scol > trow) v = NEG;              // causal mask
        s[r] = v;
      }
      sc[half] = s;
    }
    // online softmax statistics (rows live across 16-lane groups)
    float corr[8];
#pragma unroll
    for (int r = 0; r < 8; ++r) {
      float mloc = fmaxf(sc[0][r], sc[1][r]);
#pragma unroll
      for (int off = 1; off < 16; off <<= 1) mloc = fmaxf(mloc, __shfl_xor(mloc, off, 32));
      float mn = fmaxf(mrun[r], mloc);
      float p0 = __expf(sc[0][r] - mn);
      float p1 = __expf(sc[1][r] - mn);
      sc[0][r] = p0; sc[1][r] = p1;
      float ps = p0 + p1;
#pragma unroll
      for (int off = 1; off < 16; off <<= 1) ps += __shfl_xor(ps, off, 32);
      corr[r] = __expf(mrun[r] - mn);
      lrun[r] = lrun[r] * corr[r] + ps;
      mrun[r] = mn;
    }
    // stage P (16x32) as bf16 in LDS so it can be reloaded as an A-fragment
    __syncthreads();
#pragma unroll
    for (int half = 0; half < 2; ++half)
#pragma unroll
      for (int r = 0; r < 8; ++r)
        Plds[(r + 8 * g) * 32 + 16 * half + c] = to_bf16(sc[half][r]);
    __syncthreads();
    bf16x16 pa = load_frag_a(Plds, 32, lane);
#pragma unroll
    for (int j = 0; j < 4; ++j) {
#pragma unroll
      for (int r = 0; r < 8; ++r) acc[j][r] *= corr[r];
      bf16x16 vb = load_frag_b(vh + (size_t)(16 * j) * TT + s0, TT, lane);
      acc[j] = wmma_bf16(pa, vb, acc[j]);
    }
  }
#pragma unroll
  for (int j = 0; j < 4; ++j) {
#pragma unroll
    for (int r = 0; r < 8; ++r) {
      int t = t0 + r + 8 * g;
      int col = h * HSZ + 16 * j + c;
      attn_out[((size_t)b * TT + t) * DD + col] = to_bf16(acc[j][r] / lrun[r]);
    }
  }
}

// ---------------------------------------------------------------------------
// GEMM + bias + residual, fp32 out:  C[M][N] = A(bf16) @ Bt^T + bias + resid
// One wave per 16x64 output tile.
// ---------------------------------------------------------------------------
__global__ void gemm_bias_res_kernel(const __bf16* __restrict__ A,
                                     const __bf16* __restrict__ Bt,
                                     const float* __restrict__ bias,
                                     const float* __restrict__ resid,
                                     float* __restrict__ Cf,
                                     int N, int K) {
  int lane = threadIdx.x;
  int n0 = blockIdx.x * 64, m0 = blockIdx.y * 16;
  f32x8 acc[4] = {};
  for (int k0 = 0; k0 < K; k0 += 32) {
    if (k0 + 32 < K) __builtin_prefetch(A + (size_t)m0 * K + k0 + 32, 0, 1);
    bf16x16 a = load_frag_a(A + (size_t)m0 * K + k0, K, lane);
#pragma unroll
    for (int j = 0; j < 4; ++j) {
      bf16x16 b = load_frag_b(Bt + (size_t)(n0 + 16 * j) * K + k0, K, lane);
      acc[j] = wmma_bf16(a, b, acc[j]);
    }
  }
  int g = lane >> 4, c = lane & 15;
#pragma unroll
  for (int j = 0; j < 4; ++j) {
    int col = n0 + 16 * j + c;
    float bv = bias[col];
#pragma unroll
    for (int r = 0; r < 8; ++r) {
      int row = m0 + r + 8 * g;
      float v = acc[j][r] + bv + resid[(size_t)row * N + col];
      Cf[(size_t)row * N + col] = v;
    }
  }
}

// ---------------------------------------------------------------------------
// Fused SwiGLU FFN1: act = (h2@W1a + b1a) * silu(h2@W1g + b1g)  -> bf16
// One wave computes both the value tile (col n) and gate tile (col n+1536).
// ---------------------------------------------------------------------------
__global__ void ffn1_swiglu_kernel(const __bf16* __restrict__ H2,
                                   const __bf16* __restrict__ W1T,
                                   const float* __restrict__ b1,
                                   __bf16* __restrict__ act) {
  int lane = threadIdx.x;
  int n0 = blockIdx.x * 64, m0 = blockIdx.y * 16;
  f32x8 accA[4] = {};
  f32x8 accG[4] = {};
  for (int k0 = 0; k0 < DD; k0 += 32) {
    bf16x16 a = load_frag_a(H2 + (size_t)m0 * DD + k0, DD, lane);
#pragma unroll
    for (int j = 0; j < 4; ++j) {
      bf16x16 bA = load_frag_b(W1T + (size_t)(n0 + 16 * j) * DD + k0, DD, lane);
      accA[j] = wmma_bf16(a, bA, accA[j]);
      bf16x16 bG = load_frag_b(W1T + (size_t)(HALFD + n0 + 16 * j) * DD + k0, DD, lane);
      accG[j] = wmma_bf16(a, bG, accG[j]);
    }
  }
  int g = lane >> 4, c = lane & 15;
#pragma unroll
  for (int j = 0; j < 4; ++j) {
    int col = n0 + 16 * j + c;
#pragma unroll
    for (int r = 0; r < 8; ++r) {
      int row = m0 + r + 8 * g;
      float av = accA[j][r] + b1[col];
      float gv = accG[j][r] + b1[HALFD + col];
      float silu = gv / (1.0f + __expf(-gv));
      act[(size_t)row * HALFD + col] = to_bf16(av * silu);
    }
  }
}

// ---------------------------------------------------------------------------
// Host-side orchestration
// ---------------------------------------------------------------------------
extern "C" void kernel_launch(void* const* d_in, const int* in_sizes, int n_in,
                              void* d_out, int out_size, void* d_ws, size_t ws_size,
                              hipStream_t stream) {
  (void)in_sizes; (void)n_in; (void)out_size; (void)ws_size;
  const float* x  = (const float*)d_in[0];
  const float* Wq = (const float*)d_in[1];
  const float* Wk = (const float*)d_in[2];
  const float* Wv = (const float*)d_in[3];
  const float* Wo = (const float*)d_in[4];
  const float* bo = (const float*)d_in[5];
  const float* W1 = (const float*)d_in[6];
  const float* b1 = (const float*)d_in[7];
  const float* W2 = (const float*)d_in[8];
  const float* b2 = (const float*)d_in[9];
  const float* g1 = (const float*)d_in[10];
  const float* g2 = (const float*)d_in[11];
  float* out = (float*)d_out;

  char* p = (char*)d_ws;
  auto take = [&](size_t bytes) -> char* {
    char* r = p;
    p += (bytes + 255) & ~(size_t)255;
    return r;
  };
  __bf16* h1   = (__bf16*)take((size_t)BT * DD * 2);
  __bf16* WqT  = (__bf16*)take((size_t)DD * DD * 2);
  __bf16* WkT  = (__bf16*)take((size_t)DD * DD * 2);
  __bf16* WvT  = (__bf16*)take((size_t)DD * DD * 2);
  __bf16* WoT  = (__bf16*)take((size_t)DD * DD * 2);
  __bf16* W1T  = (__bf16*)take((size_t)DD * FFD * 2);
  __bf16* W2T  = (__bf16*)take((size_t)HALFD * DD * 2);
  __bf16* Qb   = (__bf16*)take((size_t)BT * DD * 2);
  __bf16* Kb   = (__bf16*)take((size_t)BT * DD * 2);
  __bf16* Vb   = (__bf16*)take((size_t)BT * DD * 2);
  __bf16* attn = (__bf16*)take((size_t)BT * DD * 2);
  float*  x2   = (float*)take((size_t)BT * DD * 4);
  __bf16* act  = (__bf16*)take((size_t)BT * HALFD * 2);
  __bf16* h2   = h1;  // reuse: h1 dead after QKV

  // 1) weight casts (fp32 -> bf16, transposed to B^T layout)
  {
    int n = DD * DD;
    cast_T_qkv_kernel<<<(n + 255) / 256, 256, 0, stream>>>(Wq, WqT);
    cast_T_qkv_kernel<<<(n + 255) / 256, 256, 0, stream>>>(Wk, WkT);
    cast_T_qkv_kernel<<<(n + 255) / 256, 256, 0, stream>>>(Wv, WvT);
    cast_T_2d_kernel<<<(n + 255) / 256, 256, 0, stream>>>(Wo, WoT, DD, DD);
    long long n1 = (long long)DD * FFD;
    cast_T_2d_kernel<<<(int)((n1 + 255) / 256), 256, 0, stream>>>(W1, W1T, DD, FFD);
    long long n2 = (long long)HALFD * DD;
    cast_T_2d_kernel<<<(int)((n2 + 255) / 256), 256, 0, stream>>>(W2, W2T, HALFD, DD);
  }
  // 2) h = rmsnorm(x, g1) -> bf16
  rmsnorm_cast_kernel<<<BT, 256, 0, stream>>>(x, g1, h1);
  // 3) Q/K/V projections
  {
    dim3 grid(DD / 64, BT / 16, 3);
    qkv_kernel<<<grid, 32, 0, stream>>>(h1, WqT, WkT, WvT, Qb, Kb, Vb);
  }
  // 4) causal flash attention
  {
    dim3 grid(TT / 16, BB * HH);
    attention_kernel<<<grid, 32, 0, stream>>>(Qb, Kb, Vb, attn);
  }
  // 5) x2 = x + attn @ Wo + bo
  {
    dim3 grid(DD / 64, BT / 16);
    gemm_bias_res_kernel<<<grid, 32, 0, stream>>>(attn, WoT, bo, x, x2, DD, DD);
  }
  // 6) h2 = rmsnorm(x2, g2) -> bf16
  rmsnorm_cast_kernel<<<BT, 256, 0, stream>>>(x2, g2, h2);
  // 7) act = silu(gate) * a  (fused SwiGLU FFN1)
  {
    dim3 grid(HALFD / 64, BT / 16);
    ffn1_swiglu_kernel<<<grid, 32, 0, stream>>>(h2, W1T, b1, act);
  }
  // 8) out = x2 + act @ W2 + b2
  {
    dim3 grid(DD / 64, BT / 16);
    gemm_bias_res_kernel<<<grid, 32, 0, stream>>>(act, W2T, b2, x2, out, DD, HALFD);
  }
}
